// GlobalRouters_12979391168719
// MI455X (gfx1250) — compile-verified, matrix operand order checked by reading, not run on hardware
//
#include <hip/hip_runtime.h>
#include <math.h>

typedef __attribute__((ext_vector_type(16))) __bf16 v16bf;
typedef __attribute__((ext_vector_type(8)))  __bf16 v8bf;
typedef __attribute__((ext_vector_type(8)))  float  v8f;

#define N_USED 960   // 512 qk + 256 v + 128 rel + 64 val (know segment unused)

union FragBF {
  v16bf v;
  v8bf  h[2];
};

// ---------------------------------------------------------------------------
// Kernel A: prep — bf16 copies of proj_w and row-normalized neuron_emb,
// plus zeroing of the per-batch accumulator (must happen every launch).
// grid = 1024 (pw) + 960 (embn rows) + 120 (zero) blocks, 64 threads each.
// ---------------------------------------------------------------------------
__global__ __launch_bounds__(64)
void prep_kernel(const float* __restrict__ proj_w,
                 const float* __restrict__ neuron_emb,
                 __bf16* __restrict__ pw,
                 __bf16* __restrict__ embn,
                 float* __restrict__ wacc) {
  __shared__ float red[64];
  const int blk = blockIdx.x, tid = threadIdx.x;
  if (blk < 1024) {
    const int i = blk * 64 + tid;
    pw[i] = (__bf16)proj_w[i];
  } else if (blk < 1024 + 960) {
    const int n = blk - 1024;
    const float v = neuron_emb[n * 64 + tid];
    red[tid] = v * v;
    __syncthreads();
    for (int o = 32; o > 0; o >>= 1) {
      if (tid < o) red[tid] += red[tid + o];
      __syncthreads();
    }
    embn[n * 64 + tid] = (__bf16)(v * rsqrtf(red[0]));
  } else {
    const int i = (blk - (1024 + 960)) * 64 + tid;
    if (i < 8 * N_USED) wacc[i] = 0.0f;
  }
}

// ---------------------------------------------------------------------------
// Kernel B: fused  h = x@Wt+b  ->  logits = h@embn^T  -> segment softmax ->
// importance-weighted accumulation into wacc[b][0..959].
// grid = (8*4096)/16 = 2048 blocks of 128 threads (4 waves).
// ---------------------------------------------------------------------------
__global__ __launch_bounds__(128)
void main_kernel(const float* __restrict__ x,
                 const float* __restrict__ importance,
                 const float* __restrict__ proj_b,
                 const __bf16* __restrict__ pw,
                 const __bf16* __restrict__ embn,
                 float* __restrict__ wacc) {
  // smem aliased: phase 1-2 use it as bf16 x-tile [16][1024] (32 KB),
  // phase 3+ use it as fp32 logits [16][960] (60 KB).
  __shared__ __align__(16) unsigned char smem[16 * N_USED * 4];
  __shared__ __align__(16) __bf16 hbf[16 * 64];
  __shared__ float rowM[4][16];
  __shared__ float rowS[4][16];

  __bf16* xbf   = (__bf16*)smem;
  float*  logit = (float*)smem;

  const int tid   = threadIdx.x;
  const int lane  = tid & 31;
  const int wave  = tid >> 5;
  const int mrow  = lane & 15;          // A-fragment row / B-fragment col
  const int kh    = (lane >> 4) * 8;    // K-half offset per 16-bit WMMA layout
  const int mbase = (lane >> 4) << 3;   // C-fragment row base
  const int b     = blockIdx.x >> 8;                 // batch (256 blocks / batch)
  const int srow0 = (blockIdx.x & 255) * 16;         // s of tile row 0
  const long g0   = (long)blockIdx.x * 16;           // global row of tile

  // ---- Phase 1: stage 16x1024 x-tile as bf16 in LDS (coalesced float4) ----
  {
    const float4* xg = (const float4*)(x + g0 * 1024);
    for (int j = 0; j < 32; ++j) {
      const int i4 = tid + j * 128;          // 0..4095
      float4 v = xg[i4];
      __bf16* p = &xbf[i4 * 4];
      p[0] = (__bf16)v.x; p[1] = (__bf16)v.y;
      p[2] = (__bf16)v.z; p[3] = (__bf16)v.w;
    }
  }
  __syncthreads();

  // ---- Phase 2: h = x @ proj_w^T + b ; wave w owns ds in [16w, 16w+16) ----
  {
    v8f hacc = {0.f, 0.f, 0.f, 0.f, 0.f, 0.f, 0.f, 0.f};
    const int ds = wave * 16 + mrow;
    for (int k0 = 0; k0 < 1024; k0 += 32) {
      FragBF a, bm;
      a.h[0]  = *(const v8bf*)&xbf[mrow * 1024 + k0 + kh];
      a.h[1]  = *(const v8bf*)&xbf[mrow * 1024 + k0 + 16 + kh];
      bm.h[0] = *(const v8bf*)&pw[ds * 1024 + k0 + kh];
      bm.h[1] = *(const v8bf*)&pw[ds * 1024 + k0 + 16 + kh];
      hacc = __builtin_amdgcn_wmma_f32_16x16x32_bf16(
          false, a.v, false, bm.v, (short)0, hacc, false, false);
    }
    const float bias = proj_b[ds];
    for (int r = 0; r < 8; ++r)
      hbf[(mbase + r) * 64 + ds] = (__bf16)(hacc[r] + bias);
  }
  __syncthreads();

  // ---- Phase 3: logits = h @ embn^T ; wave w owns n in [240w, 240w+240) ----
  {
    FragBF a0, a1;
    a0.h[0] = *(const v8bf*)&hbf[mrow * 64 + 0  + kh];
    a0.h[1] = *(const v8bf*)&hbf[mrow * 64 + 16 + kh];
    a1.h[0] = *(const v8bf*)&hbf[mrow * 64 + 32 + kh];
    a1.h[1] = *(const v8bf*)&hbf[mrow * 64 + 48 + kh];
    for (int t = 0; t < 15; ++t) {
      const int n = wave * 240 + t * 16 + mrow;
      FragBF b0, b1;
      b0.h[0] = *(const v8bf*)&embn[n * 64 + 0  + kh];
      b0.h[1] = *(const v8bf*)&embn[n * 64 + 16 + kh];
      b1.h[0] = *(const v8bf*)&embn[n * 64 + 32 + kh];
      b1.h[1] = *(const v8bf*)&embn[n * 64 + 48 + kh];
      v8f acc = {0.f, 0.f, 0.f, 0.f, 0.f, 0.f, 0.f, 0.f};
      acc = __builtin_amdgcn_wmma_f32_16x16x32_bf16(
          false, a0.v, false, b0.v, (short)0, acc, false, false);
      acc = __builtin_amdgcn_wmma_f32_16x16x32_bf16(
          false, a1.v, false, b1.v, (short)0, acc, false, false);
      for (int r = 0; r < 8; ++r)
        logit[(mbase + r) * N_USED + n] = acc[r];
    }
  }
  __syncthreads();

  // ---- Phase 4: per-row per-segment max & exp-sum (8 lanes per row) ----
  {
    const int r  = tid >> 3;   // row 0..15
    const int g8 = tid & 7;    // lane within 8-group (wave32-aligned)
    const int segs[5] = {0, 512, 768, 896, 960};
    const float imp = importance[b * 4096 + srow0 + r];
    for (int sg = 0; sg < 4; ++sg) {
      float m = -3.402823466e38f;
      for (int c = segs[sg] + g8; c < segs[sg + 1]; c += 8)
        m = fmaxf(m, logit[r * N_USED + c]);
      m = fmaxf(m, __shfl_xor(m, 1, 32));
      m = fmaxf(m, __shfl_xor(m, 2, 32));
      m = fmaxf(m, __shfl_xor(m, 4, 32));
      float z = 0.f;
      for (int c = segs[sg] + g8; c < segs[sg + 1]; c += 8) {
        const float e = __expf(logit[r * N_USED + c] - m);
        z += e;
        logit[r * N_USED + c] = e;   // cache exp in place
      }
      z += __shfl_xor(z, 1, 32);
      z += __shfl_xor(z, 2, 32);
      z += __shfl_xor(z, 4, 32);
      if (g8 == 0) { rowM[sg][r] = m; rowS[sg][r] = imp / z; }
    }
  }
  __syncthreads();

  // ---- Phase 5: reduce 16 rows per column, one atomic per column/block ----
  for (int c = tid; c < N_USED; c += 128) {
    const int sg = (c < 512) ? 0 : (c < 768) ? 1 : (c < 896) ? 2 : 3;
    float part = 0.f;
    for (int rr = 0; rr < 16; ++rr)
      part += logit[rr * N_USED + c] * rowS[sg][rr];
    atomicAdd(&wacc[b * N_USED + c], part);
  }
}

// ---------------------------------------------------------------------------
// Kernel C: tiny per-batch top-k / sort / sparsify and output packing.
// Output (flat float): idx_qk[8][64] | idx_v[8][32] | rel_Q[8][128] |
//                      rel_K[8][128] | val_w[8][64]
// ---------------------------------------------------------------------------
__global__ __launch_bounds__(32)
void finalize_kernel(const float* __restrict__ wacc, float* __restrict__ out) {
  const int b = blockIdx.x;
  if (threadIdx.x != 0) return;
  const float* w = wacc + b * N_USED;
  const float NEG = -3.402823466e38f;

  {  // qk: top-64 of 512, indices sorted ascending
    float buf[512]; int idx[64];
    for (int i = 0; i < 512; ++i) buf[i] = w[i];
    for (int k = 0; k < 64; ++k) {
      int bi = 0; float bv = buf[0];
      for (int i = 1; i < 512; ++i) if (buf[i] > bv) { bv = buf[i]; bi = i; }
      idx[k] = bi; buf[bi] = NEG;
    }
    for (int i = 1; i < 64; ++i) {
      int v = idx[i], j = i - 1;
      while (j >= 0 && idx[j] > v) { idx[j + 1] = idx[j]; --j; }
      idx[j + 1] = v;
    }
    for (int k = 0; k < 64; ++k) out[b * 64 + k] = (float)idx[k];
  }
  {  // v: top-32 of 256, indices sorted ascending
    float buf[256]; int idx[32];
    for (int i = 0; i < 256; ++i) buf[i] = w[512 + i];
    for (int k = 0; k < 32; ++k) {
      int bi = 0; float bv = buf[0];
      for (int i = 1; i < 256; ++i) if (buf[i] > bv) { bv = buf[i]; bi = i; }
      idx[k] = bi; buf[bi] = NEG;
    }
    for (int i = 1; i < 32; ++i) {
      int v = idx[i], j = i - 1;
      while (j >= 0 && idx[j] > v) { idx[j + 1] = idx[j]; --j; }
      idx[j + 1] = v;
    }
    for (int k = 0; k < 32; ++k) out[512 + b * 32 + k] = (float)idx[k];
  }
  {  // rel: dense top-16 sparsify of 128 -> rel_Q and rel_K (identical)
    float buf[128], od[128];
    for (int i = 0; i < 128; ++i) { buf[i] = w[768 + i]; od[i] = 0.f; }
    for (int k = 0; k < 16; ++k) {
      int bi = 0; float bv = buf[0];
      for (int i = 1; i < 128; ++i) if (buf[i] > bv) { bv = buf[i]; bi = i; }
      od[bi] = w[768 + bi]; buf[bi] = NEG;
    }
    for (int i = 0; i < 128; ++i) {
      out[768  + b * 128 + i] = od[i];
      out[1792 + b * 128 + i] = od[i];
    }
  }
  {  // val: dense top-3 sparsify of 64
    float buf[64], od[64];
    for (int i = 0; i < 64; ++i) { buf[i] = w[896 + i]; od[i] = 0.f; }
    for (int k = 0; k < 3; ++k) {
      int bi = 0; float bv = buf[0];
      for (int i = 1; i < 64; ++i) if (buf[i] > bv) { bv = buf[i]; bi = i; }
      od[bi] = w[896 + bi]; buf[bi] = NEG;
    }
    for (int i = 0; i < 64; ++i) out[2816 + b * 64 + i] = od[i];
  }
}

// ---------------------------------------------------------------------------
extern "C" void kernel_launch(void* const* d_in, const int* in_sizes, int n_in,
                              void* d_out, int out_size, void* d_ws, size_t ws_size,
                              hipStream_t stream) {
  const float* x    = (const float*)d_in[0];   // (8,4096,1024)
  const float* imp  = (const float*)d_in[1];   // (8,4096)
  const float* pw_f = (const float*)d_in[2];   // (64,1024)
  const float* pb   = (const float*)d_in[3];   // (64,)
  const float* nemb = (const float*)d_in[4];   // (1216,64)
  float* out = (float*)d_out;

  char* ws = (char*)d_ws;
  __bf16* pw   = (__bf16*)ws;                          // 131072 B
  __bf16* embn = (__bf16*)(ws + 131072);               // 122880 B (960 rows used)
  float*  wacc = (float*)(ws + 131072 + 122880);       // 30720 B

  prep_kernel<<<1024 + 960 + 120, 64, 0, stream>>>(pw_f, nemb, pw, embn, wacc);
  main_kernel<<<2048, 128, 0, stream>>>(x, imp, pb, pw, embn, wacc);
  finalize_kernel<<<8, 32, 0, stream>>>(wacc, out);
}
